// Attention_59785944760754
// MI455X (gfx1250) — matrix operand order repaired and checked by reading, not compiled, not run on hardware
//
#include <hip/hip_runtime.h>

// ---------------------------------------------------------------------------
// Problem constants (reference: B=2, S=2048, E=768, H=12, V=50257)
// ---------------------------------------------------------------------------
#define B_  2
#define S_  2048
#define E_  768
#define H_  12
#define V_  50257
#define HE_ (H_ * E_)                 // 9216
#define RSQRT_E 0.03608439182435161f  // 1/sqrt(768)

typedef __attribute__((ext_vector_type(16))) __bf16 v16bf;
typedef __attribute__((ext_vector_type(8)))  float  v8f;
typedef __attribute__((ext_vector_type(4)))  unsigned int u32x4;
typedef __attribute__((ext_vector_type(4)))  float  f32x4;
typedef __attribute__((ext_vector_type(4)))  int    i32x4;
typedef __attribute__((ext_vector_type(8)))  int    i32x8;

union Frag16 { v16bf v; u32x4 q[2]; };
union Pack8  { u32x4 q; __bf16 h[8]; };

// ---------------------------------------------------------------------------
// WMMA fragment helpers (layouts per CDNA5 ISA 7.12.2, wave32)
//   A 16x32 bf16 : lane&15 -> row, K = {k0..k0+7, 16+k0..16+k0+7}, k0=(lane>>4)*8
//   B 32x16 bf16 : lane&15 -> column n, K = {k0..k0+7, 16+k0..16+k0+7}, k0=(lane>>4)*8
//   C/D 16x16 f32: vgpr g holds row ((lane>=16)?8:0)+g, col = lane&15
// ---------------------------------------------------------------------------
static __device__ __forceinline__ v16bf load_A_frag(const __bf16* base, int pitch) {
  const int lane = threadIdx.x & 31;
  const int row  = lane & 15;
  const int k0   = (lane >> 4) << 3;  // 0 or 8
  const __bf16* p = base + row * pitch + k0;
  Frag16 f;
  f.q[0] = *reinterpret_cast<const u32x4*>(p);
  f.q[1] = *reinterpret_cast<const u32x4*>(p + 16);
  return f.v;
}

static __device__ __forceinline__ v16bf load_B_frag(const __bf16* base, int pitch) {
  const int lane = threadIdx.x & 31;
  const int n  = lane & 15;
  const int kh = (lane >> 4) << 3;    // 0 or 8
  const __bf16* p = base + n * pitch + kh;
  Frag16 f;
  f.q[0] = *reinterpret_cast<const u32x4*>(p);
  f.q[1] = *reinterpret_cast<const u32x4*>(p + 16);
  return f.v;
}

static __device__ __forceinline__ v8f wmma_bf16f32(v16bf a, v16bf b, v8f c) {
  // (neg_a, A, neg_b, B, c_mod, C, reuse_a, reuse_b)
  return __builtin_amdgcn_wmma_f32_16x16x32_bf16(false, a, false, b, (short)0, c,
                                                 false, false);
}

// ---------------------------------------------------------------------------
// Tensor Data Mover: async 2-D tile copy global -> LDS (ISA ch.8 D# layout).
// data_size = 4B elements; pad_enable inserts (pad_amount_code+1) DWORDs of LDS
// padding every (2<<pad_interval_code) DWORDs -> implements padded tile pitch.
// Issued by one wave (EXEC ignored by TDM); completion via TENSORcnt.
// ---------------------------------------------------------------------------
static __device__ __forceinline__ unsigned lds_addr_u32(const void* p) {
  // Generic (flat) LDS addresses carry the LDS byte offset in the low 32 bits
  // (see ISA 10.2).
  return (unsigned)(unsigned long long)p;
}

#if __has_builtin(__builtin_amdgcn_tensor_load_to_lds)
#define HAVE_TDM 1
#else
#define HAVE_TDM 0
#endif

#if HAVE_TDM
static __device__ __forceinline__ void tdm_load_2d(unsigned lds_byte,
                                                   const void* gsrc,
                                                   unsigned tile_w_dw,
                                                   unsigned tile_rows,
                                                   unsigned row_stride_dw,
                                                   unsigned pad_interval_code,
                                                   unsigned pad_amount_code) {
  const unsigned long long ga = (unsigned long long)gsrc;
  u32x4 g0 = {0u, 0u, 0u, 0u};
  g0[0] = 1u;                                   // count=1, user descriptor
  g0[1] = lds_byte;                             // lds_addr (bytes)
  g0[2] = (unsigned)ga;                         // global_addr[31:0]
  g0[3] = (unsigned)((ga >> 32) & 0x01ffffffu)  // global_addr[56:32]
        | (2u << 30);                           // type = 2 ("image")
  const unsigned td0 = row_stride_dw;           // tensor_dim0 (elems per line)
  const unsigned td1 = 1u << 20;                // tensor_dim1: ample lines
  i32x8 g1 = {0, 0, 0, 0, 0, 0, 0, 0};
  g1[0] = (int)((2u << 16)                      // data_size = 4 bytes
              | (1u << 20)                      // pad_enable
              | (pad_interval_code << 22)
              | (pad_amount_code << 25));
  g1[1] = (int)((td0 & 0xffffu) << 16);
  g1[2] = (int)(((td0 >> 16) & 0xffffu) | ((td1 & 0xffffu) << 16));
  g1[3] = (int)(((td1 >> 16) & 0xffffu) | ((tile_w_dw & 0xffffu) << 16));
  g1[4] = (int)(tile_rows & 0xffffu);           // tile_dim1; tile_dim2 = 0 (2-D)
  g1[5] = (int)row_stride_dw;                   // tensor_dim0_stride[31:0]
  i32x4 z4 = {0, 0, 0, 0};
#if defined(__clang_major__) && (__clang_major__ >= 23)
  i32x8 z8 = {0, 0, 0, 0, 0, 0, 0, 0};
  __builtin_amdgcn_tensor_load_to_lds(g0, g1, z4, z4, z8, 0);
#else
  __builtin_amdgcn_tensor_load_to_lds(g0, g1, z4, z4, 0);
#endif
}
#endif  // HAVE_TDM

// ---------------------------------------------------------------------------
// Kernel 1/2: deterministic 2-stage column mean of wte [V, E]
// ---------------------------------------------------------------------------
#define WTE_CHUNKS 48
#define WTE_ROWS_PER_CHUNK 1048  // 48*1048 >= 50257

__global__ void wte_partial_kernel(const float* __restrict__ wte,
                                   float* __restrict__ part) {
  const int ch = blockIdx.x;                  // 0..47
  const int e  = blockIdx.y * 256 + threadIdx.x;
  const int v0 = ch * WTE_ROWS_PER_CHUNK;
  int v1 = v0 + WTE_ROWS_PER_CHUNK;
  if (v1 > V_) v1 = V_;
  float s = 0.0f;
  for (int v = v0; v < v1; ++v) s += wte[(size_t)v * E_ + e];
  part[ch * E_ + e] = s;
}

__global__ void wte_final_kernel(const float* __restrict__ part,
                                 float* __restrict__ avg) {
  const int e = blockIdx.x * 256 + threadIdx.x;
  float s = 0.0f;
  #pragma unroll
  for (int c = 0; c < WTE_CHUNKS; ++c) s += part[c * E_ + e];
  avg[e] = s * (1.0f / (float)V_);
}

// ---------------------------------------------------------------------------
// Kernel 3: LayerNorm -> bf16 xn  (one block per row, 256 threads, 3 elems/thr)
// ---------------------------------------------------------------------------
__global__ __launch_bounds__(256)
void ln_kernel(const float* __restrict__ x, const float* __restrict__ lnw,
               __bf16* __restrict__ xn) {
  const int row = blockIdx.x;
  const int tid = threadIdx.x;
  const float* xr = x + (size_t)row * E_;
  float v0 = xr[tid], v1 = xr[tid + 256], v2 = xr[tid + 512];
  float s  = v0 + v1 + v2;
  float s2 = v0 * v0 + v1 * v1 + v2 * v2;
  #pragma unroll
  for (int d = 1; d < 32; d <<= 1) {
    s  += __shfl_xor(s,  d, 32);
    s2 += __shfl_xor(s2, d, 32);
  }
  __shared__ float rs[8], rs2[8];
  if ((tid & 31) == 0) { rs[tid >> 5] = s; rs2[tid >> 5] = s2; }
  __syncthreads();
  s = 0.0f; s2 = 0.0f;
  #pragma unroll
  for (int w = 0; w < 8; ++w) { s += rs[w]; s2 += rs2[w]; }
  const float mu  = s * (1.0f / (float)E_);
  const float var = s2 * (1.0f / (float)E_) - mu * mu;
  const float inv = rsqrtf(var + 1e-5f);
  __bf16* o = xn + (size_t)row * E_;
  o[tid]       = (__bf16)((v0 - mu) * inv * lnw[tid]);
  o[tid + 256] = (__bf16)((v1 - mu) * inv * lnw[tid + 256]);
  o[tid + 512] = (__bf16)((v2 - mu) * inv * lnw[tid + 512]);
}

// ---------------------------------------------------------------------------
// Kernel 4: attention pass 1 — per-row logsumexp of causal logits.
//   logits = (xn .* w_h^2 / sqrt(E)) @ xn^T  via bf16 WMMA, K-dim = 768.
//   block = 128 thr (4 waves); each wave owns a 16-row strip of a 64-row tile.
//   K-tiles arrive by TDM (async DMA, padded pitch-72); Q-tiles are VALU-scaled
//   copies that overlap with the in-flight TDM.
// ---------------------------------------------------------------------------
__global__ __launch_bounds__(128)
void attn_lse_kernel(const __bf16* __restrict__ xn, const float* __restrict__ Wqk,
                     float* __restrict__ lse) {
  const int it = blockIdx.x;   // query tile of 64 rows
  const int h  = blockIdx.y;
  const int b  = blockIdx.z;
  const int tid = threadIdx.x;
  const int wave = tid >> 5;
  const int lane = tid & 31;
  const int nlc  = lane & 15;
  const int half8 = (lane >> 4) << 3;
  const int r0 = it * 64;
  const size_t xb = (size_t)b * S_ * E_;

  __shared__ __bf16 Qs[64][72];
  __shared__ __bf16 Ks[64][72];
  __shared__ float  scs[E_];

  const unsigned ksBase = lds_addr_u32(&Ks[0][0]);
  (void)ksBase;

  for (int i = tid; i < E_; i += 128) {
    float w = Wqk[h * E_ + i];
    scs[i] = w * w * RSQRT_E;
  }
  __syncthreads();

  float m_run[8], l_run[8];
  #pragma unroll
  for (int g = 0; g < 8; ++g) { m_run[g] = -3.0e38f; l_run[g] = 0.0f; }

  for (int j = 0; j <= it; ++j) {
    v8f s_frag[4];
    #pragma unroll
    for (int nj = 0; nj < 4; ++nj)
      #pragma unroll
      for (int g = 0; g < 8; ++g) s_frag[nj][g] = 0.0f;

    for (int ec2 = 0; ec2 < 12; ++ec2) {     // K-dim chunks of 64 over E=768
      const int e0 = ec2 * 64;
      __syncthreads();
#if HAVE_TDM
      if (wave == 0)                         // K tile: 64 rows x 32 DW, pad 32+4
        tdm_load_2d(ksBase, xn + xb + (size_t)(j * 64) * E_ + e0,
                    32u, 64u, (unsigned)(E_ / 2), 4u, 3u);
#endif
      #pragma unroll
      for (int c = 0; c < 4; ++c) {
        const int lin = c * 128 + tid;       // 0..511
        const int row = lin >> 3;            // 0..63
        const int grp = lin & 7;             // 8-wide column group
        Pack8 qin, qout;
        qin.q = *reinterpret_cast<const u32x4*>(
            xn + xb + (size_t)(r0 + row) * E_ + e0 + grp * 8);
        #pragma unroll
        for (int i = 0; i < 8; ++i)
          qout.h[i] = (__bf16)((float)qin.h[i] * scs[e0 + grp * 8 + i]);
        *reinterpret_cast<u32x4*>(&Qs[row][grp * 8]) = qout.q;
#if !HAVE_TDM
        Pack8 kin;
        kin.q = *reinterpret_cast<const u32x4*>(
            xn + xb + (size_t)(j * 64 + row) * E_ + e0 + grp * 8);
        *reinterpret_cast<u32x4*>(&Ks[row][grp * 8]) = kin.q;
#endif
      }
#if HAVE_TDM
      if (wave == 0) __builtin_amdgcn_s_wait_tensorcnt(0);
#endif
      __syncthreads();
      #pragma unroll
      for (int kk = 0; kk < 2; ++kk) {
        v16bf a  = load_A_frag(&Qs[wave * 16][kk * 32], 72);
        v16bf b0 = load_B_frag(&Ks[ 0][kk * 32], 72);
        v16bf b1 = load_B_frag(&Ks[16][kk * 32], 72);
        v16bf b2 = load_B_frag(&Ks[32][kk * 32], 72);
        v16bf b3 = load_B_frag(&Ks[48][kk * 32], 72);
        s_frag[0] = wmma_bf16f32(a, b0, s_frag[0]);
        s_frag[1] = wmma_bf16f32(a, b1, s_frag[1]);
        s_frag[2] = wmma_bf16f32(a, b2, s_frag[2]);
        s_frag[3] = wmma_bf16f32(a, b3, s_frag[3]);
      }
    }

    // online max/sum update (row spread across a 16-lane half per C vgpr)
    const int rowbase = r0 + wave * 16 + half8;
    const int colbase = j * 64 + nlc;
    #pragma unroll
    for (int g = 0; g < 8; ++g) {
      const int rowg = rowbase + g;
      float sv[4];
      float smax = -3.0e38f;
      #pragma unroll
      for (int nj = 0; nj < 4; ++nj) {
        float sv0 = s_frag[nj][g];
        if (colbase + nj * 16 > rowg) sv0 = -3.0e38f;   // causal mask
        sv[nj] = sv0;
        smax = fmaxf(smax, sv0);
      }
      #pragma unroll
      for (int d = 1; d < 16; d <<= 1) smax = fmaxf(smax, __shfl_xor(smax, d, 32));
      const float mnew = fmaxf(m_run[g], smax);
      float ps = 0.0f;
      #pragma unroll
      for (int nj = 0; nj < 4; ++nj) ps += __expf(sv[nj] - mnew);
      #pragma unroll
      for (int d = 1; d < 16; d <<= 1) ps += __shfl_xor(ps, d, 32);
      l_run[g] = l_run[g] * __expf(m_run[g] - mnew) + ps;
      m_run[g] = mnew;
    }
  }

  if (nlc == 0) {
    float* lrow = lse + ((size_t)b * H_ + h) * S_ + r0 + wave * 16 + half8;
    #pragma unroll
    for (int g = 0; g < 8; ++g) lrow[g] = m_run[g] + __logf(l_run[g]);
  }
}

// ---------------------------------------------------------------------------
// Kernel 5: attention pass 2 — recompute S, P = exp(S - lse), acc += P@V for a
// 256-wide output e-chunk. Vt aliases Qs/Ks in LDS (phase-disjoint), static
// LDS kept < 64 KB. K-tiles via TDM as in pass 1.
// ---------------------------------------------------------------------------
#define P2_PS   0
#define P2_QS   9216
#define P2_KS   18432
#define P2_VT   9216          // aliases Qs/Ks region, 256*72*2 = 36864 bytes
#define P2_SCS  46080
#define P2_AVG  49152
#define P2_LSE  50176
#define P2_SMEM 50432

__global__ __launch_bounds__(128)
void attn_out_kernel(const __bf16* __restrict__ xn, const float* __restrict__ Wqk,
                     const float* __restrict__ avg, const float* __restrict__ lse,
                     __bf16* __restrict__ aout) {
  const int it = blockIdx.x;
  const int h  = blockIdx.y;
  const int bz = blockIdx.z;    // b*3 + ec
  const int b  = bz / 3;
  const int ec = bz % 3;
  const int e_out0 = ec * 256;
  const int tid = threadIdx.x;
  const int wave = tid >> 5;
  const int lane = tid & 31;
  const int nlc  = lane & 15;
  const int half8 = (lane >> 4) << 3;
  const int r0 = it * 64;
  const size_t xb = (size_t)b * S_ * E_;

  __shared__ alignas(16) char smem[P2_SMEM];
  __bf16 (*Ps)[72] = reinterpret_cast<__bf16(*)[72]>(smem + P2_PS);
  __bf16 (*Qs)[72] = reinterpret_cast<__bf16(*)[72]>(smem + P2_QS);
  __bf16 (*Ks)[72] = reinterpret_cast<__bf16(*)[72]>(smem + P2_KS);
  __bf16 (*Vt)[72] = reinterpret_cast<__bf16(*)[72]>(smem + P2_VT);
  float* scs  = reinterpret_cast<float*>(smem + P2_SCS);
  float* avgs = reinterpret_cast<float*>(smem + P2_AVG);
  float* lses = reinterpret_cast<float*>(smem + P2_LSE);

  const unsigned ksBase = lds_addr_u32(&Ks[0][0]);
  (void)ksBase;

  for (int i = tid; i < E_; i += 128) {
    float w = Wqk[h * E_ + i];
    scs[i] = w * w * RSQRT_E;
  }
  for (int i = tid; i < 256; i += 128) avgs[i] = avg[e_out0 + i];
  if (tid < 64) lses[tid] = lse[((size_t)b * H_ + h) * S_ + r0 + tid];
  __syncthreads();

  float lse_r[8];
  #pragma unroll
  for (int g = 0; g < 8; ++g) lse_r[g] = lses[wave * 16 + half8 + g];

  v8f acc[16];                               // 16 rows x 256 cols per wave
  #pragma unroll
  for (int nj = 0; nj < 16; ++nj)
    #pragma unroll
    for (int g = 0; g < 8; ++g) acc[nj][g] = 0.0f;

  for (int j = 0; j <= it; ++j) {
    // ---- S tile: (xn.*w^2/sqrt(E)) @ xn^T over full K=768 ----
    v8f s_frag[4];
    #pragma unroll
    for (int nj = 0; nj < 4; ++nj)
      #pragma unroll
      for (int g = 0; g < 8; ++g) s_frag[nj][g] = 0.0f;

    for (int ec2 = 0; ec2 < 12; ++ec2) {
      const int e0 = ec2 * 64;
      __syncthreads();                       // also protects aliased Vt reuse
#if HAVE_TDM
      if (wave == 0)
        tdm_load_2d(ksBase, xn + xb + (size_t)(j * 64) * E_ + e0,
                    32u, 64u, (unsigned)(E_ / 2), 4u, 3u);
#endif
      #pragma unroll
      for (int c = 0; c < 4; ++c) {
        const int lin = c * 128 + tid;
        const int row = lin >> 3;
        const int grp = lin & 7;
        Pack8 qin, qout;
        qin.q = *reinterpret_cast<const u32x4*>(
            xn + xb + (size_t)(r0 + row) * E_ + e0 + grp * 8);
        #pragma unroll
        for (int i = 0; i < 8; ++i)
          qout.h[i] = (__bf16)((float)qin.h[i] * scs[e0 + grp * 8 + i]);
        *reinterpret_cast<u32x4*>(&Qs[row][grp * 8]) = qout.q;
#if !HAVE_TDM
        Pack8 kin;
        kin.q = *reinterpret_cast<const u32x4*>(
            xn + xb + (size_t)(j * 64 + row) * E_ + e0 + grp * 8);
        *reinterpret_cast<u32x4*>(&Ks[row][grp * 8]) = kin.q;
#endif
      }
#if HAVE_TDM
      if (wave == 0) __builtin_amdgcn_s_wait_tensorcnt(0);
#endif
      __syncthreads();
      #pragma unroll
      for (int kk = 0; kk < 2; ++kk) {
        v16bf a  = load_A_frag(&Qs[wave * 16][kk * 32], 72);
        v16bf b0 = load_B_frag(&Ks[ 0][kk * 32], 72);
        v16bf b1 = load_B_frag(&Ks[16][kk * 32], 72);
        v16bf b2 = load_B_frag(&Ks[32][kk * 32], 72);
        v16bf b3 = load_B_frag(&Ks[48][kk * 32], 72);
        s_frag[0] = wmma_bf16f32(a, b0, s_frag[0]);
        s_frag[1] = wmma_bf16f32(a, b1, s_frag[1]);
        s_frag[2] = wmma_bf16f32(a, b2, s_frag[2]);
        s_frag[3] = wmma_bf16f32(a, b3, s_frag[3]);
      }
    }

    // ---- P = exp(S - lse) with causal mask, stored bf16 (wave-local rows) ----
    const int rowb = wave * 16 + half8;
    #pragma unroll
    for (int nj = 0; nj < 4; ++nj) {
      #pragma unroll
      for (int g = 0; g < 8; ++g) {
        const int rowg = r0 + rowb + g;
        const int col  = j * 64 + nj * 16 + nlc;
        const float pv = (col > rowg) ? 0.0f : __expf(s_frag[nj][g] - lse_r[g]);
        Ps[rowb + g][nj * 16 + nlc] = (__bf16)pv;
      }
    }

    __syncthreads();                         // Qs/Ks reads done before Vt alias
    // ---- stage V^T chunk: Vt[e_local][key] = xn - avg_wte (bf16) ----
    #pragma unroll
    for (int rep = 0; rep < 16; ++rep) {
      const int lin = rep * 128 + tid;       // 0..2047
      const int key = lin >> 5;              // 0..63
      const int eg  = lin & 31;              // 0..31 (8 e's each)
      Pack8 vin;
      vin.q = *reinterpret_cast<const u32x4*>(
          xn + xb + (size_t)(j * 64 + key) * E_ + e_out0 + eg * 8);
      #pragma unroll
      for (int i = 0; i < 8; ++i)
        Vt[eg * 8 + i][key] = (__bf16)((float)vin.h[i] - avgs[eg * 8 + i]);
    }
    __syncthreads();

    // ---- acc += P @ V ----
    #pragma unroll
    for (int kk = 0; kk < 2; ++kk) {
      v16bf a = load_A_frag(&Ps[wave * 16][kk * 32], 72);
      #pragma unroll
      for (int njg = 0; njg < 4; ++njg) {
        v16bf b0 = load_B_frag(&Vt[(njg * 4 + 0) * 16][kk * 32], 72);
        v16bf b1 = load_B_frag(&Vt[(njg * 4 + 1) * 16][kk * 32], 72);
        v16bf b2 = load_B_frag(&Vt[(njg * 4 + 2) * 16][kk * 32], 72);
        v16bf b3 = load_B_frag(&Vt[(njg * 4 + 3) * 16][kk * 32], 72);
        acc[njg * 4 + 0] = wmma_bf16f32(a, b0, acc[njg * 4 + 0]);
        acc[njg * 4 + 1] = wmma_bf16f32(a, b1, acc[njg * 4 + 1]);
        acc[njg * 4 + 2] = wmma_bf16f32(a, b2, acc[njg * 4 + 2]);
        acc[njg * 4 + 3] = wmma_bf16f32(a, b3, acc[njg * 4 + 3]);
      }
    }
  }

  // ---- epilogue: N_reg scale, store bf16 to [B, S, H, E] ----
  #pragma unroll
  for (int nj = 0; nj < 16; ++nj) {
    #pragma unroll
    for (int g = 0; g < 8; ++g) {
      const int rowg = r0 + wave * 16 + half8 + g;
      const float o  = acc[nj][g] * (1.0f / (float)(rowg + 1));
      const int e    = e_out0 + nj * 16 + nlc;
      aout[(((size_t)b * S_ + rowg) * H_ + h) * E_ + e] = (__bf16)o;
    }
  }
}

// ---------------------------------------------------------------------------
// Kernel 6: out = attn_out[4096, 9216] @ W_o^T[9216, 768]  (bf16 WMMA GEMM)
//   block 256 thr (8 waves): 64x128 output tile, wave = 16x64 sub-tile.
//   X-tiles (raw bf16 copies) arrive by TDM; W_o tiles need f32->bf16 VALU.
// ---------------------------------------------------------------------------
__global__ __launch_bounds__(256)
void proj_kernel(const __bf16* __restrict__ X, const float* __restrict__ Wo,
                 float* __restrict__ out) {
  const int m0 = blockIdx.x * 64;
  const int n0 = blockIdx.y * 128;
  const int tid = threadIdx.x;
  const int wave = tid >> 5;
  const int lane = tid & 31;
  const int nlc  = lane & 15;
  const int half8 = (lane >> 4) << 3;
  const int mrow = (wave & 3) * 16;
  const int ncb  = (wave >> 2) * 64;

  __shared__ __bf16 Xs[64][40];
  __shared__ __bf16 Ws[128][40];
  const unsigned xsBase = lds_addr_u32(&Xs[0][0]);
  (void)xsBase;

  v8f acc[4];
  #pragma unroll
  for (int nj = 0; nj < 4; ++nj)
    #pragma unroll
    for (int g = 0; g < 8; ++g) acc[nj][g] = 0.0f;

  for (int k0 = 0; k0 < HE_; k0 += 32) {
    __syncthreads();
#if HAVE_TDM
    if (wave == 0)                           // X tile: 64 rows x 16 DW, pad 16+4
      tdm_load_2d(xsBase, X + (size_t)m0 * HE_ + k0,
                  16u, 64u, (unsigned)(HE_ / 2), 3u, 3u);
#else
    {
      const int row = tid >> 2;              // 0..63
      const int grp = tid & 3;               // 8-wide group
      *reinterpret_cast<u32x4*>(&Xs[row][grp * 8]) =
          *reinterpret_cast<const u32x4*>(X + (size_t)(m0 + row) * HE_ + k0 + grp * 8);
    }
#endif
    #pragma unroll
    for (int rep = 0; rep < 2; ++rep) {
      const int lin = rep * 256 + tid;       // 0..511
      const int n   = lin >> 2;              // 0..127
      const int grp = lin & 3;
      const float* wp = Wo + (size_t)(n0 + n) * HE_ + k0 + grp * 8;
      const f32x4 f0 = *reinterpret_cast<const f32x4*>(wp);
      const f32x4 f1 = *reinterpret_cast<const f32x4*>(wp + 4);
      Pack8 pk;
      #pragma unroll
      for (int i = 0; i < 4; ++i) {
        pk.h[i]     = (__bf16)f0[i];
        pk.h[4 + i] = (__bf16)f1[i];
      }
      *reinterpret_cast<u32x4*>(&Ws[n][grp * 8]) = pk.q;
    }
#if HAVE_TDM
    if (wave == 0) __builtin_amdgcn_s_wait_tensorcnt(0);
#endif
    __syncthreads();
    v16bf a  = load_A_frag(&Xs[mrow][0], 40);
    v16bf b0 = load_B_frag(&Ws[ncb +  0][0], 40);
    v16bf b1 = load_B_frag(&Ws[ncb + 16][0], 40);
    v16bf b2 = load_B_frag(&Ws[ncb + 32][0], 40);
    v16bf b3 = load_B_frag(&Ws[ncb + 48][0], 40);
    acc[0] = wmma_bf16f32(a, b0, acc[0]);
    acc[1] = wmma_bf16f32(a, b1, acc[1]);
    acc[2] = wmma_bf16f32(a, b2, acc[2]);
    acc[3] = wmma_bf16f32(a, b3, acc[3]);
  }

  #pragma unroll
  for (int nj = 0; nj < 4; ++nj) {
    #pragma unroll
    for (int g = 0; g < 8; ++g) {
      const int m = m0 + mrow + half8 + g;
      const int n = n0 + ncb + nj * 16 + nlc;
      out[(size_t)m * E_ + n] = acc[nj][g];
    }
  }
}

// ---------------------------------------------------------------------------
// Launch. Workspace layout (bytes, 256-aligned):
//   xn   bf16 [B,S,E]      @ 0          (6,291,456)
//   lse  f32  [B,H,S]      @ 6,291,456  (196,608)
//   aout bf16 [B,S,H,E]    @ 6,488,064  (75,497,472)
//   part f32  [48,E]       @ 81,985,536 (147,456)
//   avg  f32  [E]          @ 82,132,992 (3,072)      total ~82.1 MB
// ---------------------------------------------------------------------------
extern "C" void kernel_launch(void* const* d_in, const int* in_sizes, int n_in,
                              void* d_out, int out_size, void* d_ws, size_t ws_size,
                              hipStream_t stream) {
  (void)in_sizes; (void)n_in; (void)out_size; (void)ws_size;
  const float* x   = (const float*)d_in[0];
  // d_in[1] = e, d_in[2] = p : unused by the reference forward path
  const float* lnw = (const float*)d_in[3];
  const float* Wqk = (const float*)d_in[4];
  const float* Wo  = (const float*)d_in[5];
  const float* wte = (const float*)d_in[6];
  float* out = (float*)d_out;

  char* ws = (char*)d_ws;
  __bf16* xn   = (__bf16*)(ws + 0);
  float*  lse  = (float*) (ws + 6291456);
  __bf16* aout = (__bf16*)(ws + 6488064);
  float*  part = (float*) (ws + 81985536);
  float*  avg  = (float*) (ws + 82132992);

  wte_partial_kernel<<<dim3(WTE_CHUNKS, 3), dim3(256), 0, stream>>>(wte, part);
  wte_final_kernel<<<dim3(3), dim3(256), 0, stream>>>(part, avg);
  ln_kernel<<<dim3(B_ * S_), dim3(256), 0, stream>>>(x, lnw, xn);
  attn_lse_kernel<<<dim3(S_ / 64, H_, B_), dim3(128), 0, stream>>>(xn, Wqk, lse);
  attn_out_kernel<<<dim3(S_ / 64, H_, B_ * 3), dim3(128), 0, stream>>>(xn, Wqk, avg,
                                                                       lse, aout);
  proj_kernel<<<dim3((B_ * S_) / 64, E_ / 128), dim3(256), 0, stream>>>(aout, Wo, out);
}
